// EntmaxAttention_73624329388481
// MI455X (gfx1250) — matrix operand order repaired and checked by reading, not compiled, not run on hardware
//
#include <hip/hip_runtime.h>
#include <hip/hip_bf16.h>

// Problem sizes (fixed by the reference)
#define BB 64
#define SS 4096
#define HH 256
#define AA 128

typedef __attribute__((ext_vector_type(16))) __bf16 v16bf;
typedef __attribute__((ext_vector_type(8)))  float  v8f;

union FragU { uint4 u[2]; v16bf v; };

__device__ __forceinline__ unsigned short f2bf(float f) {
    // round-to-nearest-even fp32 -> bf16
    unsigned int u = __float_as_uint(f);
    unsigned int r = u + 0x7FFFu + ((u >> 16) & 1u);
    return (unsigned short)(r >> 16);
}

// ---------------------------------------------------------------------------
// K0: W1 [H][A] fp32 -> W1T [A][H] bf16 (so each WMMA B-column is contiguous)
// ---------------------------------------------------------------------------
__global__ void prep_w1t(const float* __restrict__ W1, unsigned short* __restrict__ W1T) {
    int idx = blockIdx.x * blockDim.x + threadIdx.x;   // 0 .. A*H-1
    int k = idx & (HH - 1);
    int n = idx >> 8;                                  // H == 256
    W1T[n * HH + k] = f2bf(W1[k * AA + n]);
}

// ---------------------------------------------------------------------------
// K1: fused  scores[m] = b2 + sum_n W2[n] * tanh( (X W1)[m,n] + b1[n] )
// One block = 16 rows (M-tile). 8 waves, wave w owns N-tile [16w,16w+16).
// K loop: 8 iterations of v_wmma_f32_16x16x32_bf16.  `h` never hits memory.
// ---------------------------------------------------------------------------
__global__ void __launch_bounds__(256) gemm_score(
    const float* __restrict__ lstm, const unsigned short* __restrict__ W1T,
    const float* __restrict__ b1, const float* __restrict__ W2,
    const float* __restrict__ b2, float* __restrict__ scores)
{
    __shared__ __align__(16) unsigned short Arows[16][HH]; // 16 rows of X, bf16 (8 KB)
    __shared__ float sc_w[8][16];                          // per-wave partial scores

    const int t  = threadIdx.x;
    const int m0 = blockIdx.x * 16;

    // Stage 16 rows of lstm_output as bf16 (coalesced: thread t handles column t)
    for (int i = 0; i < 16; ++i)
        Arows[i][t] = f2bf(lstm[(size_t)(m0 + i) * HH + t]);
    __syncthreads();

    const int w    = t >> 5;        // wave id 0..7  == N-tile
    const int lane = t & 31;
    const int col  = lane & 15;     // tile row (A) / tile col (B,C)
    const int hh   = lane >> 4;     // lane half
    const int n    = w * 16 + col;  // global output column, 0..127

    v8f acc = {};
    const unsigned short* brow = W1T + n * HH;
    for (int kt = 0; kt < HH; kt += 32) {
        FragU afr, bfr;
        // A 16x32 bf16 layout: lanes<16: K=0..7 & 16..23 ; lanes>=16: K=8..15 & 24..31
        afr.u[0] = *(const uint4*)&Arows[col][kt + hh * 8];
        afr.u[1] = *(const uint4*)&Arows[col][kt + 16 + hh * 8];
        // B 32x16 bf16 layout: lanes<16: K=0..7 & 16..23 of column n ; lanes>=16: K=8..15 & 24..31
        bfr.u[0] = *(const uint4*)&brow[kt + hh * 8];
        bfr.u[1] = *(const uint4*)&brow[kt + 16 + hh * 8];
        acc = __builtin_amdgcn_wmma_f32_16x16x32_bf16(
            false, afr.v, false, bfr.v, (short)0, acc, false, false);
    }

    // Epilogue: C layout -> lanes<16: rows 0..7 in acc[0..7]; lanes>=16: rows 8..15
    const float b1n = b1[n];
    const float w2n = W2[n];
    for (int r = 0; r < 8; ++r) {
        float p = tanhf(acc[r] + b1n) * w2n;
        p += __shfl_xor(p, 1, 32);
        p += __shfl_xor(p, 2, 32);
        p += __shfl_xor(p, 4, 32);
        p += __shfl_xor(p, 8, 32);          // reduce over the 16-lane column group
        if (col == 0) sc_w[w][hh * 8 + r] = p;
    }
    __syncthreads();
    if (t < 16) {
        float s = b2[0];
        for (int ww = 0; ww < 8; ++ww) s += sc_w[ww][t];  // deterministic combine
        scores[m0 + t] = s;
    }
}

// ---------------------------------------------------------------------------
// K2: exact entmax-1.5 per batch row (one 1024-thread block per b).
// Bitonic sort of 4096 in LDS, dual prefix scan (sum, sum-of-squares),
// support count, tau*, then weights = clip(z - tau*)^2.
// ---------------------------------------------------------------------------
__global__ void __launch_bounds__(1024) entmax_kernel(
    const float* __restrict__ scores, float* __restrict__ weights)
{
    __shared__ float zs[SS];       // 16 KB sorted values
    __shared__ float ps1[1024];
    __shared__ float ps2[1024];
    __shared__ float red[1024];
    __shared__ float s_tau;
    __shared__ int   kcnt;

    const int t = threadIdx.x;
    const int b = blockIdx.x;
    const float* sr = scores + (size_t)b * SS;

    // z = score/2, find max
    float zl[4];
    float mloc = -3.402823466e38f;
    for (int j = 0; j < 4; ++j) {
        float z = sr[t * 4 + j] * 0.5f;
        zl[j] = z;
        mloc = fmaxf(mloc, z);
    }
    red[t] = mloc;
    __syncthreads();
    for (int s = 512; s > 0; s >>= 1) {
        if (t < s) red[t] = fmaxf(red[t], red[t + s]);
        __syncthreads();
    }
    const float mx = red[0];
    for (int j = 0; j < 4; ++j) zs[t * 4 + j] = zl[j] - mx;
    if (t == 0) kcnt = 0;
    __syncthreads();

    // bitonic sort, descending
    for (unsigned k = 2; k <= SS; k <<= 1) {
        for (unsigned j = k >> 1; j > 0; j >>= 1) {
            for (unsigned i = t; i < SS; i += 1024) {
                unsigned p = i ^ j;
                if (p > i) {
                    float a = zs[i], c = zs[p];
                    bool up = ((i & k) == 0);
                    if (up ? (a < c) : (a > c)) { zs[i] = c; zs[p] = a; }
                }
            }
            __syncthreads();
        }
    }

    // per-thread chunk (4 contiguous sorted values) partial sums
    float lt1 = 0.0f, lt2 = 0.0f;
    float c0[4];
    for (int j = 0; j < 4; ++j) {
        float v = zs[t * 4 + j];
        c0[j] = v;
        lt1 += v; lt2 += v * v;
    }
    ps1[t] = lt1; ps2[t] = lt2;
    __syncthreads();
    // Hillis-Steele inclusive scan over 1024 chunk totals
    for (int off = 1; off < 1024; off <<= 1) {
        float a1 = 0.0f, a2 = 0.0f;
        if (t >= off) { a1 = ps1[t - off]; a2 = ps2[t - off]; }
        float c1 = ps1[t], c2 = ps2[t];
        __syncthreads();
        ps1[t] = c1 + a1; ps2[t] = c2 + a2;
        __syncthreads();
    }
    float r1 = ps1[t] - lt1;   // exclusive prefix (sum)
    float r2 = ps2[t] - lt2;   // exclusive prefix (sum of squares)
    float taus[4];
    int cnt = 0;
    for (int j = 0; j < 4; ++j) {
        float v = c0[j];
        r1 += v; r2 += v * v;
        float kk    = (float)(t * 4 + j + 1);
        float mean  = r1 / kk;
        float msq   = r2 / kk;
        float ssv   = kk * (msq - mean * mean);
        float delta = fmaxf((1.0f - ssv) / kk, 0.0f);
        float tau   = mean - sqrtf(delta);
        taus[j] = tau;
        if (tau <= v) cnt++;
    }
    atomicAdd(&kcnt, cnt);                 // integer atomic: deterministic
    __syncthreads();
    int idx = kcnt - 1;                    // k_star - 1  (k_star >= 1 always)
    if (idx >= t * 4 && idx < t * 4 + 4) s_tau = taus[idx - t * 4];
    __syncthreads();
    const float tau_star = s_tau;

    float* wr = weights + (size_t)b * SS;
    for (int j = 0; j < 4; ++j) {
        int i = t * 4 + j;
        float z  = sr[i] * 0.5f - mx;
        float wv = fmaxf(z - tau_star, 0.0f);
        wr[i] = wv * wv;
    }
}

// ---------------------------------------------------------------------------
// K3/K4: context[b,h] = sum_s X[b,s,h] * w[b,s]  (deterministic 2-stage)
// ---------------------------------------------------------------------------
__global__ void __launch_bounds__(256) ctx_partial(
    const float* __restrict__ lstm, const float* __restrict__ weights,
    float* __restrict__ partials)
{
    __shared__ float wl[256];
    const int t = threadIdx.x;          // h index
    const int b = blockIdx.x;
    const int c = blockIdx.y;           // S-chunk 0..15
    const int s0 = c * 256;
    wl[t] = weights[(size_t)b * SS + s0 + t];
    __syncthreads();
    float acc = 0.0f;
    const float* base = lstm + ((size_t)b * SS + s0) * HH + t;
    for (int s = 0; s < 256; ++s) acc += base[(size_t)s * HH] * wl[s];
    partials[((size_t)b * 16 + c) * HH + t] = acc;
}

__global__ void __launch_bounds__(256) ctx_reduce(
    const float* __restrict__ partials, float* __restrict__ ctx)
{
    const int t = threadIdx.x;
    const int b = blockIdx.x;
    float s = 0.0f;
    for (int c = 0; c < 16; ++c) s += partials[((size_t)b * 16 + c) * HH + t];
    ctx[b * HH + t] = s;
}

// ---------------------------------------------------------------------------
extern "C" void kernel_launch(void* const* d_in, const int* in_sizes, int n_in,
                              void* d_out, int out_size, void* d_ws, size_t ws_size,
                              hipStream_t stream) {
    (void)in_sizes; (void)n_in; (void)out_size; (void)ws_size;
    const float* lstm = (const float*)d_in[0];   // [B,S,H] f32
    const float* W1   = (const float*)d_in[1];   // [H,A]
    const float* b1   = (const float*)d_in[2];   // [A]
    const float* W2   = (const float*)d_in[3];   // [A,1]
    const float* b2   = (const float*)d_in[4];   // [1]

    char* ws = (char*)d_ws;
    unsigned short* W1T = (unsigned short*)ws;                         // 64 KB bf16 W1^T
    float* scores   = (float*)(ws + 65536);                            // 1 MB  [B*S]
    float* partials = (float*)(ws + 65536 + (size_t)BB * SS * 4);      // 1 MB  [B*16*H]

    float* ctx     = (float*)d_out;              // context first: [B,H]
    float* weights = (float*)d_out + BB * HH;    // then weights:  [B,S]

    prep_w1t   <<<(HH * AA) / 256, 256, 0, stream>>>(W1, W1T);
    gemm_score <<<(BB * SS) / 16,  256, 0, stream>>>(lstm, W1T, b1, W2, b2, scores);
    entmax_kernel<<<BB, 1024, 0, stream>>>(scores, weights);
    ctx_partial<<<dim3(BB, 16), 256, 0, stream>>>(lstm, weights, partials);
    ctx_reduce <<<BB, 256, 0, stream>>>(partials, ctx);
}
